// M3GNet_48258252538051
// MI455X (gfx1250) — matrix operand order repaired and checked by reading, not compiled, not run on hardware
//
#include <hip/hip_runtime.h>
#include <math.h>

typedef __attribute__((ext_vector_type(16))) _Float16 v16h;
typedef __attribute__((ext_vector_type(8)))  float    v8f;

static constexpr int   G   = 32;
static constexpr int   APG = 100, EPG = 5000, TPG = 31250;
static constexpr int   A   = G * APG;      // 3200
static constexpr int   E   = G * EPG;      // 160000
static constexpr int   T   = G * TPG;      // 1000000
static constexpr int   F   = 128, NB = 4, NR = 4, NL = 4, LN = 16;
static constexpr float CUTF  = 5.0f;
static constexpr float CUT3F = 4.0f;
static constexpr float PIF   = 3.14159265358979323846f;

// ---------------------------------------------------------------- helpers
__device__ __forceinline__ float sigmoidf_(float x) { return 1.0f / (1.0f + __expf(-x)); }
__device__ __forceinline__ float swishf_(float x)   { return x * sigmoidf_(x); }

__device__ __forceinline__ float cutoff_poly_(float t) {
    if (t < 1.0f) {
        float t2 = t * t, t3 = t2 * t, t4 = t3 * t, t5 = t4 * t;
        return 1.0f - 10.0f * t3 + 15.0f * t4 - 6.0f * t5;
    }
    return 0.0f;
}

__device__ __forceinline__ void bessel4_(float r, float c, float* out) {
    float t  = r / c;
    float fc = cutoff_poly_(t);
    float s  = sqrtf(2.0f / c);
    #pragma unroll
    for (int n = 0; n < NR; ++n)
        out[n] = s * __sinf((float)(n + 1) * PIF * t) / r * fc;
}

__device__ __forceinline__ int find_seg_(int idx, const int* cum) {
    int g = 0;
    while (g < G - 1 && idx >= cum[g + 1]) ++g;
    return g;
}

// ---------------------------------------------------------------- cumsums
__global__ void k_cum(const int* tne, const int* tnang, const int* tnat,
                      int* ecum, int* tcum, int* acum) {
    if (threadIdx.x == 0) {
        ecum[0] = 0; tcum[0] = 0; acum[0] = 0;
        for (int g = 0; g < G; ++g) {
            ecum[g + 1] = ecum[g] + tne[g];
            tcum[g + 1] = tcum[g] + tnang[g];
            acum[g + 1] = acum[g] + tnat[g];
        }
    }
}

// ------------------------------------------- pack W1..W4 into B-fragment order
// packed[((b*4+mat)*384*128) + (((step*8+ntile)*32+lane)*16 + h)]
//   = W[mat][b][k*128 + n],  k = step*32 + (lane>>4)*16 + h,  n = ntile*16 + (lane&15)
__global__ void k_packW(const float* W1, const float* W2, const float* W3, const float* W4,
                        _Float16* packed) {
    int tid = blockIdx.x * blockDim.x + threadIdx.x;
    const int total = NB * 4 * 3 * F * F;   // 786432
    if (tid >= total) return;
    int h    = tid & 15;
    int t1   = tid >> 4;
    int lane = t1 & 31;
    int t2   = t1 >> 5;
    int nt   = t2 & 7;
    int t3   = t2 >> 3;
    int step = t3 % 12;
    int bm   = t3 / 12;
    int b    = bm >> 2;
    int mat  = bm & 3;
    int k = step * 32 + ((lane >> 4) << 4) + h;
    int n = nt * 16 + (lane & 15);
    const float* W = (mat == 0) ? W1 : (mat == 1) ? W2 : (mat == 2) ? W3 : W4;
    packed[tid] = (_Float16)W[(size_t)b * 3 * F * F + k * F + n];
}

// ---------------------------------------------------------------- edge geometry
__global__ void k_edges(const float* pos, const float* cell, const float* eoff_in,
                        const int* eidx, const int* ecum,
                        float* edge_vec, float* edge_dist, float* rbf0) {
    int e = blockIdx.x * blockDim.x + threadIdx.x;
    if (e >= E) return;
    int g = find_seg_(e, ecum);
    const float* c = cell + g * 9;
    float o0 = eoff_in[e * 3 + 0], o1 = eoff_in[e * 3 + 1], o2 = eoff_in[e * 3 + 2];
    float f0 = o0 * c[0] + o1 * c[3] + o2 * c[6];
    float f1 = o0 * c[1] + o1 * c[4] + o2 * c[7];
    float f2 = o0 * c[2] + o1 * c[5] + o2 * c[8];
    int i = eidx[e], j = eidx[E + e];
    float v0 = pos[j * 3 + 0] - pos[i * 3 + 0] + f0;
    float v1 = pos[j * 3 + 1] - pos[i * 3 + 1] + f1;
    float v2 = pos[j * 3 + 2] - pos[i * 3 + 2] + f2;
    float d = sqrtf(v0 * v0 + v1 * v1 + v2 * v2);
    edge_vec[e * 3 + 0] = v0; edge_vec[e * 3 + 1] = v1; edge_vec[e * 3 + 2] = v2;
    edge_dist[e] = d;
    float bes[NR];
    bessel4_(d, CUTF, bes);
    #pragma unroll
    for (int n = 0; n < NR; ++n) rbf0[e * NR + n] = bes[n];
}

// edge = swish(rbf0 @ W_enc + b_enc)
__global__ void k_edge_enc(const float* rbf0, const float* Wenc, const float* benc,
                           float* edge) {
    int tid = blockIdx.x * blockDim.x + threadIdx.x;
    if (tid >= E * F) return;
    int e = tid >> 7, f = tid & 127;
    float s = benc[f];
    #pragma unroll
    for (int n = 0; n < NR; ++n) s += rbf0[e * NR + n] * Wenc[n * F + f];
    edge[tid] = swishf_(s);
}

__global__ void k_atom_init(const int* anum, const float* emb, float* atom) {
    int tid = blockIdx.x * blockDim.x + threadIdx.x;
    if (tid >= A * F) return;
    int a = tid >> 7, f = tid & 127;
    atom[tid] = emb[anum[a] * F + f];
}

__global__ void k_zero(float* p, int n) {
    int tid = blockIdx.x * blockDim.x + threadIdx.x;
    if (tid < n) p[tid] = 0.0f;
}

// w = sigmoid(atom @ Wt + bt)
__global__ void k_w(const float* atom, const float* Wt, const float* bt, float* w) {
    int tid = blockIdx.x * blockDim.x + threadIdx.x;
    if (tid >= A * LN) return;
    int a = tid >> 4, j = tid & 15;
    float s = bt[j];
    for (int k = 0; k < F; ++k) s += atom[a * F + k] * Wt[k * LN + j];
    w[tid] = sigmoidf_(s);
}

// fused: recompute angle features, gate by w[atom_k], scatter into agg[ij]
__global__ void k_angle(const float* edge_vec, const float* edge_dist,
                        const int* tbi, const int* eidx,
                        const int* ecum, const int* tcum,
                        const float* w, float* agg) {
    int t = blockIdx.x * blockDim.x + threadIdx.x;
    if (t >= T) return;
    int g  = find_seg_(t, tcum);
    int eo = ecum[g];
    int ij = tbi[2 * t]     + eo;
    int ik = tbi[2 * t + 1] + eo;
    float a0 = edge_vec[ij * 3 + 0], a1 = edge_vec[ij * 3 + 1], a2 = edge_vec[ij * 3 + 2];
    float b0 = edge_vec[ik * 3 + 0], b1 = edge_vec[ik * 3 + 1], b2 = edge_vec[ik * 3 + 2];
    float nij = sqrtf(a0 * a0 + a1 * a1 + a2 * a2);
    float nik = sqrtf(b0 * b0 + b1 * b1 + b2 * b2);
    float cosang = (a0 * b0 + a1 * b1 + a2 * b2) / (nij * nik);
    const float eps = 1.1920929e-07f;
    cosang = fminf(fmaxf(cosang, -1.0f + eps), 1.0f - eps);
    float dik = edge_dist[ik];
    float bes[NR];
    bessel4_(dik, CUT3F, bes);
    float leg[NL];
    leg[0] = 1.0f;
    leg[1] = cosang;
    leg[2] = 0.5f * (3.0f * cosang * cosang - 1.0f);
    leg[3] = 0.5f * (5.0f * cosang * cosang * cosang - 3.0f * cosang);
    const float* wr = w + eidx[E + ik] * LN;   // atom_k = ej[ik]
    #pragma unroll
    for (int r = 0; r < NR; ++r)
        #pragma unroll
        for (int l = 0; l < NL; ++l)
            atomicAdd(&agg[ij * LN + r * NL + l], bes[r] * leg[l] * wr[r * NL + l]);
}

// edge += swish(agg @ Wp + bp)
__global__ void k_proj(const float* agg, const float* Wp, const float* bp, float* edge) {
    int tid = blockIdx.x * blockDim.x + threadIdx.x;
    if (tid >= E * F) return;
    int e = tid >> 7, f = tid & 127;
    float s = bp[f];
    #pragma unroll
    for (int k = 0; k < LN; ++k) s += agg[e * LN + k] * Wp[k * F + f];
    edge[tid] += swishf_(s);
}

// ---------------------------------------------------------- fused gated-MLP (WMMA)
// tile = 16 edges; 8 waves each own a 16-col slice, 4 accumulators (W1..W4).
#define XS_STRIDE 392
__global__ __launch_bounds__(256) void k_gemm(
        const float* __restrict__ atom_cur, float* edge,
        const float* __restrict__ rbf0, const int* __restrict__ eidx,
        const _Float16* __restrict__ wp1, const _Float16* __restrict__ wp2,
        const _Float16* __restrict__ wp3, const _Float16* __restrict__ wp4,
        const float* __restrict__ b1, const float* __restrict__ b2,
        const float* __restrict__ b3, const float* __restrict__ b4,
        const float* __restrict__ We, const float* __restrict__ Wa,
        float* atom_next) {
    __shared__ _Float16 xs[16 * XS_STRIDE];
    const int ebase = blockIdx.x * 16;
    const int tid   = threadIdx.x;

    // stage x = [atom[ei] | atom[ej] | edge] as f16, row-major 16 x 384
    for (int idx = tid; idx < 16 * 384; idx += 256) {
        int row = idx / 384, col = idx - row * 384;
        int e = ebase + row;
        float v;
        if (col < F)            v = atom_cur[eidx[e] * F + col];
        else if (col < 2 * F)   v = atom_cur[eidx[E + e] * F + (col - F)];
        else                    v = edge[e * F + (col - 2 * F)];
        xs[row * XS_STRIDE + col] = (_Float16)v;
    }
    __syncthreads();

    const int lane = tid & 31;
    const int wv   = tid >> 5;          // wave id -> column tile
    const int m    = lane & 15;
    const int hb   = lane >> 4;         // lane-half selects K sub-range
    v8f acc1 = {}, acc2 = {}, acc3 = {}, acc4 = {};

    for (int step = 0; step < 12; ++step) {
        const int kk = step * 32;
        // A fragment, contiguous k order: element s -> k = 16*hb + s (same as packed B)
        v16h a;
        const _Float16* xr = &xs[m * XS_STRIDE + kk];
        #pragma unroll
        for (int q = 0; q < 8; ++q) {
            int k = (hb << 4) + (q << 1);
            a[2 * q]     = xr[k];
            a[2 * q + 1] = xr[k + 1];
        }
        // B fragments: pre-packed, one contiguous 32B load per weight
        size_t boff = (size_t)((step * 8 + wv) * 32 + lane) * 16;
        v16h bf1 = *(const v16h*)(wp1 + boff);
        v16h bf2 = *(const v16h*)(wp2 + boff);
        v16h bf3 = *(const v16h*)(wp3 + boff);
        v16h bf4 = *(const v16h*)(wp4 + boff);
        acc1 = __builtin_amdgcn_wmma_f32_16x16x32_f16(false, a, false, bf1, (short)0, acc1, false, false);
        acc2 = __builtin_amdgcn_wmma_f32_16x16x32_f16(false, a, false, bf2, (short)0, acc2, false, false);
        acc3 = __builtin_amdgcn_wmma_f32_16x16x32_f16(false, a, false, bf3, (short)0, acc3, false, false);
        acc4 = __builtin_amdgcn_wmma_f32_16x16x32_f16(false, a, false, bf4, (short)0, acc4, false, false);
    }

    // epilogue: gates + rbf factors; edge in place, messages into atom_next
    const int f = wv * 16 + (lane & 15);
    float wef[NR], waf[NR];
    #pragma unroll
    for (int n = 0; n < NR; ++n) { wef[n] = We[n * F + f]; waf[n] = Wa[n * F + f]; }
    const float bb1 = b1[f], bb2 = b2[f], bb3 = b3[f], bb4 = b4[f];
    #pragma unroll
    for (int r = 0; r < 8; ++r) {
        int e = ebase + r + 8 * hb;     // D layout: VGPR r -> row r (+8 for hi lanes)
        float r0 = rbf0[e * NR + 0], r1 = rbf0[e * NR + 1],
              r2 = rbf0[e * NR + 2], r3 = rbf0[e * NR + 3];
        float rwe = r0 * wef[0] + r1 * wef[1] + r2 * wef[2] + r3 * wef[3];
        float rwa = r0 * waf[0] + r1 * waf[1] + r2 * waf[2] + r3 * waf[3];
        float ev  = edge[e * F + f];
        edge[e * F + f] = ev + swishf_(acc1[r] + bb1) * sigmoidf_(acc2[r] + bb2) * rwe;
        float msg = swishf_(acc3[r] + bb3) * sigmoidf_(acc4[r] + bb4) * rwa;
        atomicAdd(&atom_next[eidx[e] * F + f], msg);
    }
}

// ---------------------------------------------------------------- final energy head
__global__ __launch_bounds__(128) void k_energy(
        const float* atom, const float* W1, const float* b1,
        const float* W2, const float* b2, const float* W3, const float* b3,
        const int* anum, const float* scale, const float* mean,
        const int* acum, float* out) {
    __shared__ float sa[F], sh1[F], red[F];
    int a = blockIdx.x, t = threadIdx.x;
    sa[t] = atom[a * F + t];
    __syncthreads();
    float s = b1[t];
    for (int k = 0; k < F; ++k) s += sa[k] * W1[k * F + t];
    sh1[t] = swishf_(s);
    __syncthreads();
    s = b2[t];
    for (int k = 0; k < F; ++k) s += sh1[k] * W2[k * F + t];
    red[t] = swishf_(s) * W3[t];
    __syncthreads();
    for (int st = 64; st > 0; st >>= 1) {
        if (t < st) red[t] += red[t + st];
        __syncthreads();
    }
    if (t == 0) {
        float epa = red[0] + b3[0];
        int z = anum[a];
        epa = epa * scale[z] + mean[z];
        int g = find_seg_(a, acum);
        atomicAdd(&out[g], epa);
    }
}

// ---------------------------------------------------------------- host launcher
extern "C" void kernel_launch(void* const* d_in, const int* in_sizes, int n_in,
                              void* d_out, int out_size, void* d_ws, size_t ws_size,
                              hipStream_t stream) {
    const float* pos   = (const float*)d_in[0];
    const float* cell  = (const float*)d_in[1];
    const float* eoff  = (const float*)d_in[2];
    const int*   anum  = (const int*)d_in[3];
    const int*   eidx  = (const int*)d_in[4];
    const int*   tbi   = (const int*)d_in[5];
    const int*   tne   = (const int*)d_in[6];
    const int*   tnang = (const int*)d_in[7];
    const int*   tnat  = (const int*)d_in[8];
    const float* emb   = (const float*)d_in[9];
    const float* Wenc  = (const float*)d_in[10];
    const float* benc  = (const float*)d_in[11];
    const float* Wt    = (const float*)d_in[12];
    const float* bt    = (const float*)d_in[13];
    const float* Wp    = (const float*)d_in[14];
    const float* bp    = (const float*)d_in[15];
    const float* W1    = (const float*)d_in[16];
    const float* b1    = (const float*)d_in[17];
    const float* W2    = (const float*)d_in[18];
    const float* b2    = (const float*)d_in[19];
    const float* W3    = (const float*)d_in[20];
    const float* b3    = (const float*)d_in[21];
    const float* W4    = (const float*)d_in[22];
    const float* b4    = (const float*)d_in[23];
    const float* We    = (const float*)d_in[24];
    const float* Wa    = (const float*)d_in[25];
    const float* enW1  = (const float*)d_in[26];
    const float* enb1  = (const float*)d_in[27];
    const float* enW2  = (const float*)d_in[28];
    const float* enb2  = (const float*)d_in[29];
    const float* enW3  = (const float*)d_in[30];
    const float* enb3  = (const float*)d_in[31];
    const float* scl   = (const float*)d_in[32];
    const float* mn    = (const float*)d_in[33];
    float* out = (float*)d_out;

    // workspace carve-up (256B aligned)
    char* p = (char*)d_ws;
    auto carve = [&](size_t bytes) {
        void* r = (void*)p;
        p += (bytes + 255) & ~(size_t)255;
        return r;
    };
    int*      ecum      = (int*)carve((G + 1) * sizeof(int));
    int*      tcum      = (int*)carve((G + 1) * sizeof(int));
    int*      acum      = (int*)carve((G + 1) * sizeof(int));
    float*    edge_vec  = (float*)carve((size_t)E * 3 * sizeof(float));
    float*    edge_dist = (float*)carve((size_t)E * sizeof(float));
    float*    rbf0      = (float*)carve((size_t)E * NR * sizeof(float));
    float*    edge      = (float*)carve((size_t)E * F * sizeof(float));
    float*    wbuf      = (float*)carve((size_t)A * LN * sizeof(float));
    float*    agg       = (float*)carve((size_t)E * LN * sizeof(float));
    float*    atomA     = (float*)carve((size_t)A * F * sizeof(float));
    float*    atomB     = (float*)carve((size_t)A * F * sizeof(float));
    _Float16* w16       = (_Float16*)carve((size_t)NB * 4 * 3 * F * F * sizeof(_Float16));

    k_cum<<<1, 32, 0, stream>>>(tne, tnang, tnat, ecum, tcum, acum);
    {
        int total = NB * 4 * 3 * F * F;
        k_packW<<<(total + 255) / 256, 256, 0, stream>>>(W1, W2, W3, W4, w16);
    }
    k_edges<<<(E + 127) / 128, 128, 0, stream>>>(pos, cell, eoff, eidx, ecum,
                                                 edge_vec, edge_dist, rbf0);
    k_edge_enc<<<(E * F + 255) / 256, 256, 0, stream>>>(rbf0, Wenc, benc, edge);
    k_atom_init<<<(A * F + 255) / 256, 256, 0, stream>>>(anum, emb, atomA);

    float* cur = atomA;
    float* nxt = atomB;
    for (int b = 0; b < NB; ++b) {
        k_w<<<(A * LN + 255) / 256, 256, 0, stream>>>(cur, Wt + b * F * LN, bt + b * LN, wbuf);
        k_zero<<<(E * LN + 255) / 256, 256, 0, stream>>>(agg, E * LN);
        k_angle<<<(T + 255) / 256, 256, 0, stream>>>(edge_vec, edge_dist, tbi, eidx,
                                                     ecum, tcum, wbuf, agg);
        k_proj<<<(E * F + 255) / 256, 256, 0, stream>>>(agg, Wp + b * LN * F, bp + b * F, edge);
        hipMemcpyAsync(nxt, cur, (size_t)A * F * sizeof(float),
                       hipMemcpyDeviceToDevice, stream);
        const _Float16* wpb = w16 + (size_t)b * 4 * 3 * F * F;
        k_gemm<<<E / 16, 256, 0, stream>>>(cur, edge, rbf0, eidx,
                                           wpb, wpb + 3 * F * F, wpb + 2 * 3 * F * F, wpb + 3 * 3 * F * F,
                                           b1 + b * F, b2 + b * F, b3 + b * F, b4 + b * F,
                                           We + b * NR * F, Wa + b * NR * F, nxt);
        float* tmp = cur; cur = nxt; nxt = tmp;
    }
    k_zero<<<1, 32, 0, stream>>>(out, G);
    k_energy<<<A, 128, 0, stream>>>(cur, enW1, enb1, enW2, enb2, enW3, enb3,
                                    anum, scl, mn, acum, out);
}
